// CausalSparseCache_13529146982870
// MI455X (gfx1250) — matrix operand order repaired and checked by reading, not compiled, not run on hardware
//
#include <hip/hip_runtime.h>
#include <hip/hip_bf16.h>
#include <math.h>

// Problem constants (reference: B=32, T=2048, D=4096, K=16)
#define B_  32
#define T_  2048
#define D_  4096
#define K_  16
#define NEG_INF (-3.402823466e38f)

typedef __attribute__((ext_vector_type(16))) __bf16 bfv16;
typedef __attribute__((ext_vector_type(8)))  float  v8f;

union BfPack {
    bfv16 v;
    uint4 u[2];
    unsigned short s[16];
};

__device__ __forceinline__ unsigned short f2bf(float f) {
    // round-to-nearest-even float32 -> bf16 bits
    unsigned int u = __float_as_uint(f);
    unsigned int r = (u + 0x7FFFu + ((u >> 16) & 1u)) >> 16;
    return (unsigned short)r;
}

// ---------------------------------------------------------------------------
// Kernel 1: row-sum of p_all over D (monotonic proxy for mean) -> p_scalar[B*T]
// One wave (32 lanes) per row; float4 coalesced loads; wave shuffle reduce.
// This is the dominant cost: 1 GiB stream ~= 46 us at 23.3 TB/s.
// ---------------------------------------------------------------------------
__global__ void csc_rowsum_kernel(const float* __restrict__ p_all,
                                  float* __restrict__ p_scalar) {
    const int row  = blockIdx.x * 8 + (threadIdx.x >> 5);   // 8 waves/block
    const int lane = threadIdx.x & 31;
    const float4* base = (const float4*)(p_all + (size_t)row * D_);
    float s = 0.0f;
#pragma unroll 4
    for (int j = 0; j < D_ / 4 / 32; ++j) {                 // 32 iters
        float4 v = base[lane + j * 32];
        s += v.x + v.y + v.z + v.w;
    }
#pragma unroll
    for (int off = 16; off > 0; off >>= 1) s += __shfl_xor(s, off, 32);
    if (lane == 0) p_scalar[row] = s;
}

// ---------------------------------------------------------------------------
// Kernel 2: top-16 per batch via 16 rounds of masked argmax in LDS.
// One block (256 threads) per batch.
// ---------------------------------------------------------------------------
__global__ void csc_topk_kernel(const float* __restrict__ p_scalar,
                                int* __restrict__ topk_idx) {
    const int b = blockIdx.x;
    __shared__ float vals[T_];
    __shared__ float rv[256];
    __shared__ int   ri[256];
    for (int i = threadIdx.x; i < T_; i += 256) vals[i] = p_scalar[(size_t)b * T_ + i];
    __syncthreads();
    for (int it = 0; it < K_; ++it) {
        float best = NEG_INF; int bi = 0x7FFFFFFF;
        for (int i = threadIdx.x; i < T_; i += 256) {
            float v = vals[i];
            if (v > best || (v == best && i < bi)) { best = v; bi = i; }
        }
        rv[threadIdx.x] = best; ri[threadIdx.x] = bi;
        __syncthreads();
        for (int sN = 128; sN > 0; sN >>= 1) {
            if (threadIdx.x < sN) {
                float ov = rv[threadIdx.x + sN]; int oi = ri[threadIdx.x + sN];
                if (ov > rv[threadIdx.x] || (ov == rv[threadIdx.x] && oi < ri[threadIdx.x])) {
                    rv[threadIdx.x] = ov; ri[threadIdx.x] = oi;
                }
            }
            __syncthreads();
        }
        if (threadIdx.x == 0) {
            topk_idx[b * K_ + it] = ri[0];
            vals[ri[0]] = NEG_INF;
        }
        __syncthreads();
    }
}

// ---------------------------------------------------------------------------
// Kernel 3: gather top-k hidden rows + h_mean rows; convert fp32 -> bf16.
// A layout: rows 0..511 = h_topk (b*16+k), rows 512..543 = h_mean[b].
// ---------------------------------------------------------------------------
__global__ void csc_gather_bf16_kernel(const float* __restrict__ h_all,
                                       const float* __restrict__ h_mean,
                                       const int* __restrict__ topk_idx,
                                       unsigned short* __restrict__ Abf) {
    const int row = blockIdx.x; // 0..543
    const float* src;
    if (row < B_ * K_) {
        const int b = row >> 4, k = row & 15;
        src = h_all + ((size_t)b * T_ + topk_idx[b * K_ + k]) * D_;
    } else {
        src = h_mean + (size_t)(row - B_ * K_) * D_;
    }
    unsigned short* dst = Abf + (size_t)row * D_;
    for (int d = threadIdx.x; d < D_; d += 256) dst[d] = f2bf(src[d]);
}

// ---------------------------------------------------------------------------
// Kernel 4: bf16 WMMA GEMM  out[m, e] = sum_d A[m, d] * W[e, d] + bias[e]
// blockIdx.x = 16-wide N(e) stripe; the block sweeps ALL M tiles so each
// weight element is read from HBM exactly once (A is L2-resident, 4.25 MB).
//
// 3-stage pipeline, K-loop unrolled x2 so all buffers swap roles statically:
//   step i issues:  raw W fp32 loads for step i+2 (registers),
//                   cvt+ds_store of step i+1's W stripe (LDS double buffer),
//                   A-fragment loads for step i+1 (register double buffer),
//                   4 back-to-back WMMAs for step i,
//                   barrier, then B-fragment ds_load for step i+1.
// Every global load gets a full K-step of latency; no divergence anywhere
// near the WMMAs (EXEC all-1s by construction, all threads do EPT W elems).
// ---------------------------------------------------------------------------
template <int TPW, int NTHR>
__global__ __launch_bounds__(NTHR)
void csc_gemm_wmma_kernel(const unsigned short* __restrict__ Abf,
                          const float* __restrict__ W,
                          const float* __restrict__ bias,
                          float* __restrict__ out) {
    constexpr int EPT = (16 * 32) / NTHR;    // W elements per thread (2 or 8)
    const int n0   = blockIdx.x * 16;
    const int wave = threadIdx.x >> 5;
    const int lane = threadIdx.x & 31;

    // Two 16x32 bf16 stripes of W^T; row stride 56 ushort = 112 B
    // (16B-aligned rows, bank-spread across the 16 rows).
    __shared__ __attribute__((aligned(16))) unsigned short Bsh[2][16 * 56];

    v8f acc[TPW];
#pragma unroll
    for (int t = 0; t < TPW; ++t) acc[t] = (v8f){0.f,0.f,0.f,0.f,0.f,0.f,0.f,0.f};

    const int bn  = lane & 15;               // B column / A row-in-tile
    const int bk2 = (lane < 16) ? 0 : 8;     // B K-base in LDS row
    const int akb = (lane < 16) ? 0 : 8;     // A first-segment K-base

    // Per-tile A row base (row-major bf16, 16B-aligned segment starts).
    const unsigned short* arow0 =
        Abf + (size_t)(wave * TPW * 16 + bn) * D_ + akb;

    // Per-thread W source pointer and LDS targets (hoisted out of K loop).
    const int we = threadIdx.x * EPT;        // flat element in 16x32 stripe
    const int wr = we >> 5, wc = we & 31;    // stripe row (e), col (d)
    const float* wbase = W + (size_t)(n0 + wr) * D_ + wc;
    unsigned short* lds0 = &Bsh[0][wr * 56 + wc];
    unsigned short* lds1 = &Bsh[1][wr * 56 + wc];

    auto wload = [&](int d0, float* wreg) {
#pragma unroll
        for (int i = 0; i < EPT; i += 2) {
            const float2 v = *(const float2*)(wbase + d0 + i);
            wreg[i] = v.x; wreg[i + 1] = v.y;
        }
    };
    auto wstore = [&](const float* wreg, unsigned short* lp) {
#pragma unroll
        for (int i = 0; i < EPT; i += 2) {
            const unsigned int pk =
                (unsigned int)f2bf(wreg[i]) | ((unsigned int)f2bf(wreg[i + 1]) << 16);
            *(unsigned int*)(lp + i) = pk;
        }
    };
    auto loadA = [&](int d0, BfPack* ap) {
#pragma unroll
        for (int t = 0; t < TPW; ++t) {
            const unsigned short* ar = arow0 + (size_t)t * 16 * D_ + d0;
            ap[t].u[0] = *(const uint4*)(ar);
            ap[t].u[1] = *(const uint4*)(ar + 16);
        }
    };
    auto bload = [&](int buf, BfPack& bp) {
        bp.u[0] = *(const uint4*)&Bsh[buf][bn * 56 + bk2];
        bp.u[1] = *(const uint4*)&Bsh[buf][bn * 56 + bk2 + 16];
    };
    auto domma = [&](const BfPack* ap, const BfPack& bp) {
#pragma unroll
        for (int t = 0; t < TPW; ++t)
            acc[t] = __builtin_amdgcn_wmma_f32_16x16x32_bf16(
                false, ap[t].v, false, bp.v, (short)0, acc[t], false, false);
    };
    auto cl = [](int d) { return (d < D_) ? d : d - D_; };  // wrap (redundant tail)

    float  wregA[EPT], wregB[EPT];
    BfPack apA[TPW], apB[TPW];
    BfPack bpA, bpB;

    // Prologue: stripe(0) -> LDS buf0; stripe(32) in flight in wregA; A(0); bp(0).
    wload(0, wregB);
    wstore(wregB, lds0);
    wload(32, wregA);
    loadA(0, apA);
    __syncthreads();
    bload(0, bpA);

    // 128 K-steps of 32, two per iteration with static buffer-role swap.
    for (int d0 = 0; d0 < D_; d0 += 64) {
        // step i (d0): compute buf0; stage W(d0+32)->buf1; fetch W(d0+64), A(d0+32)
        wload(cl(d0 + 64), wregB);
        wstore(wregA, lds1);
        loadA(d0 + 32, apB);
        domma(apA, bpA);
        __syncthreads();
        bload(1, bpB);

        // step i+1 (d0+32): compute buf1; stage W(d0+64)->buf0; fetch W(d0+96), A(d0+64)
        wload(cl(d0 + 96), wregA);
        wstore(wregB, lds0);
        loadA(cl(d0 + 64), apA);
        domma(apB, bpB);
        __syncthreads();
        bload(0, bpA);
    }

    // Epilogue: bias add + store (lanes 0..15 cover 64 B contiguous per row).
    const int   colg = n0 + bn;
    const float bc   = bias[colg];
#pragma unroll
    for (int t = 0; t < TPW; ++t) {
        const int rbase = (wave * TPW + t) * 16 + ((lane < 16) ? 0 : 8);
#pragma unroll
        for (int r = 0; r < 8; ++r)
            out[(size_t)(rbase + r) * D_ + colg] = acc[t][r] + bc;
    }
}

// ---------------------------------------------------------------------------
// Kernel 5: scores = q.k/sqrt(D); softmax over 16; gate; residual output.
// One block (256 threads = 8 waves) per batch. Waves split the 17 dot jobs
// (16 score dots + 1 gate dot), lane-strided coalesced loads, shuffle reduce.
// ---------------------------------------------------------------------------
__global__ void csc_finalize_kernel(const float* __restrict__ h_mean,
                                    const float* __restrict__ Wg,
                                    const float* __restrict__ bg,
                                    const float* __restrict__ keys,
                                    const float* __restrict__ values,
                                    const float* __restrict__ query,
                                    float* __restrict__ out) {
    const int b    = blockIdx.x;
    const int wave = threadIdx.x >> 5;
    const int lane = threadIdx.x & 31;
    __shared__ float sc[17];
    __shared__ float attn[K_];
    __shared__ float gsh;

    const float* q  = query  + (size_t)b * D_;
    const float* hm = h_mean + (size_t)b * D_;

    for (int job = wave; job < 17; job += 8) {
        const float* x = (job < 16) ? q : hm;
        const float* y = (job < 16) ? (keys + ((size_t)b * K_ + job) * D_) : Wg;
        float s = 0.0f;
        for (int d = lane; d < D_; d += 32) s += x[d] * y[d];
#pragma unroll
        for (int off = 16; off > 0; off >>= 1) s += __shfl_xor(s, off, 32);
        if (lane == 0) sc[job] = s;
    }
    __syncthreads();

    if (threadIdx.x == 0) {
        const float inv_sqrt_d = 0.015625f; // 1/sqrt(4096)
        float mx = NEG_INF;
        for (int k = 0; k < K_; ++k) { sc[k] *= inv_sqrt_d; mx = fmaxf(mx, sc[k]); }
        float sum = 0.0f;
        for (int k = 0; k < K_; ++k) { float e = __expf(sc[k] - mx); attn[k] = e; sum += e; }
        const float inv = 1.0f / sum;
        for (int k = 0; k < K_; ++k) attn[k] *= inv;
        gsh = 1.0f / (1.0f + __expf(-(sc[16] + bg[0])));
    }
    __syncthreads();

    const float g = gsh;
    for (int d = threadIdx.x; d < D_; d += 256) {
        float s = 0.0f;
#pragma unroll
        for (int k = 0; k < K_; ++k)
            s += attn[k] * values[((size_t)b * K_ + k) * D_ + d];
        out[(size_t)b * D_ + d] = hm[d] + g * s;
    }
}

// ---------------------------------------------------------------------------
// Host launcher
// ---------------------------------------------------------------------------
extern "C" void kernel_launch(void* const* d_in, const int* in_sizes, int n_in,
                              void* d_out, int out_size, void* d_ws, size_t ws_size,
                              hipStream_t stream) {
    const float* h_mean = (const float*)d_in[0];
    const float* h_all  = (const float*)d_in[1];
    const float* p_all  = (const float*)d_in[2];
    const float* Wk     = (const float*)d_in[3];
    const float* bk     = (const float*)d_in[4];
    const float* Wv     = (const float*)d_in[5];
    const float* bv     = (const float*)d_in[6];
    const float* Wq     = (const float*)d_in[7];
    const float* bq     = (const float*)d_in[8];
    const float* Wg     = (const float*)d_in[9];
    const float* bg     = (const float*)d_in[10];
    float* out = (float*)d_out;

    // Workspace carve-up (~22.5 MB total)
    char* w = (char*)d_ws;
    float*          p_scalar = (float*)(w + 0x0);                  // 256 KB
    int*            topk     = (int*)(w + 0x40000);                // 2 KB
    unsigned short* Abf      = (unsigned short*)(w + 0x41000);     // 4.25 MB  [544][4096] bf16
    float*          keys     = (float*)(w + 0x500000);             // 8 MB     [512][4096]
    float*          valuesb  = (float*)(w + 0xD00000);             // 8 MB     [512][4096]
    float*          query    = (float*)(w + 0x1500000);            // 512 KB   [32][4096]

    // 1. p_scalar[b,t] = sum_d p_all  (monotonic in the mean -> same top-k)
    csc_rowsum_kernel<<<(B_ * T_) / 8, 256, 0, stream>>>(p_all, p_scalar);

    // 2. top-16 indices per batch
    csc_topk_kernel<<<B_, 256, 0, stream>>>(p_scalar, topk);

    // 3. gather h_topk rows + h_mean rows -> bf16 A matrix [544][4096]
    csc_gather_bf16_kernel<<<B_ * K_ + B_, 256, 0, stream>>>(h_all, h_mean, topk, Abf);

    // 4. keys / values: [512,4096] x [4096,4096]^T ; query: [32,4096] x Wq^T
    //    TPW=4 -> 8 waves cover 32 M-tiles; TPW=1 -> 2 waves cover 2 M-tiles.
    csc_gemm_wmma_kernel<4, 256><<<D_ / 16, 256, 0, stream>>>(Abf, Wk, bk, keys);
    csc_gemm_wmma_kernel<4, 256><<<D_ / 16, 256, 0, stream>>>(Abf, Wv, bv, valuesb);
    csc_gemm_wmma_kernel<1, 64><<<D_ / 16, 64, 0, stream>>>(Abf + (size_t)(B_ * K_) * D_,
                                                            Wq, bq, query);

    // 5. scores -> softmax -> gated residual output
    csc_finalize_kernel<<<B_, 256, 0, stream>>>(h_mean, Wg, bg, keys, valuesb, query, out);
}
